// LSTM_single_bounded_70626442215807
// MI455X (gfx1250) — matrix operand order repaired and checked by reading, not compiled, not run on hardware
//
#include <hip/hip_runtime.h>

// ---------------------------------------------------------------------------
// 2-layer LSTM (H=50) + linear head, fused per-batch-tile recurrence.
// Each block owns MT=32 batch rows and runs all T=512 steps of both layers
// with weights/state resident in LDS; gate GEMMs use v_wmma_f32_16x16x32_f16.
// ---------------------------------------------------------------------------

typedef __attribute__((ext_vector_type(16))) _Float16 v16h;
typedef __attribute__((ext_vector_type(8)))  float    v8f;

#define B_TOT 4096
#define T_TOT 512
#define H     50
#define G     200              // 4*H gate width
#define NT    13               // N tiles of 16  (200 -> 208 padded)
#define NP    (NT * 16)        // 208
#define KP    64               // padded K (50 -> 64), 2 k-chunks of 32
#define MT    32               // batch rows per block (2 waves x 16 rows)
#define NTHREADS 64
#define TCH   64               // x timestep chunk staged in LDS

// ---- LDS layout (bytes); every region is 64B aligned by construction ------
#define SZ_WFRAG (NT * 2 * 32 * 16 * 2)   // 26624 B : 26 B-fragments, f16
#define OFF_W0   0
#define OFF_W1A  (OFF_W0  + SZ_WFRAG)
#define OFF_W1B  (OFF_W1A + SZ_WFRAG)
#define OFF_GB   (OFF_W1B + SZ_WFRAG)
#define SZ_GB    (MT * NP * 4)            // 26624 B : gate pre-activations f32
#define OFF_XS   (OFF_GB  + SZ_GB)
#define SZ_XS    (MT * TCH * 4)           //  8192 B : staged x chunk
#define OFF_H0F  (OFF_XS  + SZ_XS)
#define SZ_HF    (2 * 2 * 32 * 16 * 2)    //  4096 B : A-fragments (2 waves x 2 kc)
#define OFF_H1F  (OFF_H0F + SZ_HF)
#define OFF_H1P  (OFF_H1F + SZ_HF)
#define SZ_H1P   (MT * H * 4)             //  6400 B : plain h1 copy for the head
#define OFF_B0   (OFF_H1P + SZ_H1P)
#define SZ_BIAS  (NP * 4)                 //   832 B
#define OFF_B1   (OFF_B0  + SZ_BIAS)
#define OFF_WI0  (OFF_B1  + SZ_BIAS)
#define SMEM_BYTES (OFF_WI0 + SZ_BIAS)    // 131776 B  (< 320KB WGP LDS)

__device__ __forceinline__ float fast_sigmoid(float x) {
    return __builtin_amdgcn_rcpf(1.0f + __expf(-x));
}
__device__ __forceinline__ float fast_tanh(float x) {
    float e = __expf(2.0f * x);
    return 1.0f - 2.0f * __builtin_amdgcn_rcpf(e + 1.0f);
}
__device__ __forceinline__ v8f wmma16(v16h a, v16h b, v8f c) {
    // D = A(16x32 f16) x B(32x16 f16) + C(16x16 f32)
    return __builtin_amdgcn_wmma_f32_16x16x32_f16(false, a, false, b,
                                                  (short)0, c, false, false);
}

// Convert a [G][H] row-major f32 weight into per-lane WMMA B-fragment layout:
// fragment (nt,kc), lane l, element e  <->  B^T[k][n] = W[n][k]
//   k = kc*32 + (e>>3)*16 + (l>>4)*8 + (e&7) ,  n = nt*16 + (l&15)     (ISA 7.12.2, 16b B)
__device__ __forceinline__ void fill_wfrag(_Float16* frag, const float* W, int tid) {
    for (int idx = tid; idx < NT * 2 * 512; idx += NTHREADS) {
        int f  = idx >> 9;            // fragment id = nt*2 + kc
        int r  = idx & 511;
        int l  = r >> 4;              // lane
        int e  = r & 15;              // element within lane
        int kc = f & 1;
        int nt = f >> 1;
        int k  = kc * 32 + ((e >> 3) << 4) + ((l >> 4) << 3) + (e & 7);
        int n  = nt * 16 + (l & 15);
        float v = (k < H && n < G) ? W[n * H + k] : 0.0f;
        frag[idx] = (_Float16)v;
    }
}

__global__ __launch_bounds__(NTHREADS)
void lstm2_wmma_kernel(const float* __restrict__ x,
                       const float* __restrict__ W_ih0, const float* __restrict__ W_hh0,
                       const float* __restrict__ b_ih0, const float* __restrict__ b_hh0,
                       const float* __restrict__ W_ih1, const float* __restrict__ W_hh1,
                       const float* __restrict__ b_ih1, const float* __restrict__ b_hh1,
                       const float* __restrict__ W_fc,  const float* __restrict__ b_fc,
                       float* __restrict__ out) {
    extern __shared__ char smem[];
    _Float16* w0f = (_Float16*)(smem + OFF_W0);   // Whh0^T fragments
    _Float16* w1a = (_Float16*)(smem + OFF_W1A);  // Wih1^T fragments
    _Float16* w1b = (_Float16*)(smem + OFF_W1B);  // Whh1^T fragments
    float*    gb  = (float*)   (smem + OFF_GB);   // [MT][NP] gate pre-acts
    float*    xs  = (float*)   (smem + OFF_XS);   // [MT][TCH]
    _Float16* h0f = (_Float16*)(smem + OFF_H0F);  // h0 A-fragments
    _Float16* h1f = (_Float16*)(smem + OFF_H1F);  // h1 A-fragments
    float*    h1p = (float*)   (smem + OFF_H1P);  // [MT][H]
    float*    b0s = (float*)   (smem + OFF_B0);   // bih0+bhh0
    float*    b1s = (float*)   (smem + OFF_B1);   // bih1+bhh1
    float*    wi0 = (float*)   (smem + OFF_WI0);  // W_ih0 column (rank-1 input)

    const int tid  = threadIdx.x;
    const int lane = tid & 31;
    const int wv   = tid >> 5;         // wave id: owns rows [16*wv, 16*wv+16)
    const int cl   = lane & 15;
    const int hiL  = lane >> 4;
    const int rb   = blockIdx.x * MT;  // global batch row base

    // ---------------- one-time init -------------------------------------
    fill_wfrag(w0f, W_hh0, tid);
    fill_wfrag(w1a, W_ih1, tid);
    fill_wfrag(w1b, W_hh1, tid);
    for (int j = tid; j < G; j += NTHREADS) {
        b0s[j] = b_ih0[j] + b_hh0[j];
        b1s[j] = b_ih1[j] + b_hh1[j];
        wi0[j] = W_ih0[j];                       // W_ih0 is [4H,1]
    }
    for (int idx = tid; idx < SZ_HF / 2; idx += NTHREADS) {
        h0f[idx] = (_Float16)0.0f;               // zero state + K padding
        h1f[idx] = (_Float16)0.0f;
    }
    float c0r[(MT * H) / NTHREADS];              // layer-0 cell state (regs)
    float c1r[(MT * H) / NTHREADS];              // layer-1 cell state (regs)
#pragma unroll
    for (int it = 0; it < (MT * H) / NTHREADS; ++it) { c0r[it] = 0.0f; c1r[it] = 0.0f; }
    __syncthreads();

    // ---------------- time loop -----------------------------------------
    for (int t = 0; t < T_TOT; ++t) {
        if ((t & (TCH - 1)) == 0) {              // stage next x chunk
            for (int idx = tid; idx < MT * TCH; idx += NTHREADS) {
                int m = idx >> 6, toff = idx & (TCH - 1);
                xs[m * TCH + toff] = x[(rb + m) * T_TOT + t + toff];
            }
            if (t + TCH < T_TOT)
                __builtin_prefetch(&x[(rb + (tid >> 1)) * T_TOT + t + TCH], 0, 1);
            __syncthreads();
        }

        // ---- layer 0 GEMM: gates0_pre = h0 @ Whh0^T  (26 WMMA / wave) ----
        {
            const v16h a0 = *(const v16h*)(h0f + ((wv * 2 + 0) * 32 + lane) * 16);
            const v16h a1 = *(const v16h*)(h0f + ((wv * 2 + 1) * 32 + lane) * 16);
#pragma unroll
            for (int nt = 0; nt < NT; ++nt) {
                v8f acc = {};
                const v16h bA = *(const v16h*)(w0f + ((nt * 2 + 0) * 32 + lane) * 16);
                const v16h bB = *(const v16h*)(w0f + ((nt * 2 + 1) * 32 + lane) * 16);
                acc = wmma16(a0, bA, acc);
                acc = wmma16(a1, bB, acc);
                const int col = nt * 16 + cl;
#pragma unroll
                for (int r = 0; r < 8; ++r)       // C layout: VGPR r -> M=r(+8)
                    gb[(wv * 16 + hiL * 8 + r) * NP + col] = acc[r];
            }
        }
        __syncthreads();

        // ---- layer 0 cell update (adds rank-1 x-term + bias) -------------
#pragma unroll
        for (int it = 0; it < (MT * H) / NTHREADS; ++it) {
            int idx = tid + it * NTHREADS;
            int m = idx / H, hp = idx - m * H;
            float xv = xs[m * TCH + (t & (TCH - 1))];
            float gi = gb[m * NP + hp        ] + xv * wi0[hp        ] + b0s[hp        ];
            float gf = gb[m * NP + hp +     H] + xv * wi0[hp +     H] + b0s[hp +     H];
            float gg = gb[m * NP + hp + 2 * H] + xv * wi0[hp + 2 * H] + b0s[hp + 2 * H];
            float go = gb[m * NP + hp + 3 * H] + xv * wi0[hp + 3 * H] + b0s[hp + 3 * H];
            float iv = fast_sigmoid(gi), fv = fast_sigmoid(gf);
            float gv = fast_tanh(gg),    ov = fast_sigmoid(go);
            float cn = fv * c0r[it] + iv * gv;
            c0r[it] = cn;
            float hn = ov * fast_tanh(cn);
            // scatter into A-fragment layout (ISA 7.12.2, 16b A 16x32)
            int kc  = hp >> 5;
            int kk  = hp & 31;
            int fhi = (kk >> 3) & 1;
            int j   = (kk < 16) ? (kk & 7) : (8 + (kk & 7));
            h0f[(((m >> 4) * 2 + kc) * 32 + fhi * 16 + (m & 15)) * 16 + j] = (_Float16)hn;
        }
        __syncthreads();

        // ---- layer 1 GEMM: h0_new@Wih1^T + h1@Whh1^T (52 WMMA / wave) ----
        {
            const v16h a0 = *(const v16h*)(h0f + ((wv * 2 + 0) * 32 + lane) * 16);
            const v16h a1 = *(const v16h*)(h0f + ((wv * 2 + 1) * 32 + lane) * 16);
            const v16h r0 = *(const v16h*)(h1f + ((wv * 2 + 0) * 32 + lane) * 16);
            const v16h r1 = *(const v16h*)(h1f + ((wv * 2 + 1) * 32 + lane) * 16);
#pragma unroll
            for (int nt = 0; nt < NT; ++nt) {
                v8f acc = {};
                acc = wmma16(a0, *(const v16h*)(w1a + ((nt * 2 + 0) * 32 + lane) * 16), acc);
                acc = wmma16(a1, *(const v16h*)(w1a + ((nt * 2 + 1) * 32 + lane) * 16), acc);
                acc = wmma16(r0, *(const v16h*)(w1b + ((nt * 2 + 0) * 32 + lane) * 16), acc);
                acc = wmma16(r1, *(const v16h*)(w1b + ((nt * 2 + 1) * 32 + lane) * 16), acc);
                const int col = nt * 16 + cl;
#pragma unroll
                for (int r = 0; r < 8; ++r)
                    gb[(wv * 16 + hiL * 8 + r) * NP + col] = acc[r];
            }
        }
        __syncthreads();

        // ---- layer 1 cell update ----------------------------------------
#pragma unroll
        for (int it = 0; it < (MT * H) / NTHREADS; ++it) {
            int idx = tid + it * NTHREADS;
            int m = idx / H, hp = idx - m * H;
            float gi = gb[m * NP + hp        ] + b1s[hp        ];
            float gf = gb[m * NP + hp +     H] + b1s[hp +     H];
            float gg = gb[m * NP + hp + 2 * H] + b1s[hp + 2 * H];
            float go = gb[m * NP + hp + 3 * H] + b1s[hp + 3 * H];
            float iv = fast_sigmoid(gi), fv = fast_sigmoid(gf);
            float gv = fast_tanh(gg),    ov = fast_sigmoid(go);
            float cn = fv * c1r[it] + iv * gv;
            c1r[it] = cn;
            float hn = ov * fast_tanh(cn);
            int kc  = hp >> 5;
            int kk  = hp & 31;
            int fhi = (kk >> 3) & 1;
            int j   = (kk < 16) ? (kk & 7) : (8 + (kk & 7));
            h1f[(((m >> 4) * 2 + kc) * 32 + fhi * 16 + (m & 15)) * 16 + j] = (_Float16)hn;
            h1p[m * H + hp] = hn;                // plain copy for the head
        }
        __syncthreads();
    }

    // ---------------- linear head on final h1 ---------------------------
    for (int idx = tid; idx < MT * 4; idx += NTHREADS) {
        int m = idx >> 2, o = idx & 3;
        float s = b_fc[o];
        for (int hh = 0; hh < H; ++hh)
            s += h1p[m * H + hh] * W_fc[o * H + hh];
        out[(rb + m) * 4 + o] = s;
    }
}

extern "C" void kernel_launch(void* const* d_in, const int* in_sizes, int n_in,
                              void* d_out, int out_size, void* d_ws, size_t ws_size,
                              hipStream_t stream) {
    (void)in_sizes; (void)n_in; (void)out_size; (void)d_ws; (void)ws_size;
    const float* x     = (const float*)d_in[0];
    const float* W_ih0 = (const float*)d_in[1];
    const float* W_hh0 = (const float*)d_in[2];
    const float* b_ih0 = (const float*)d_in[3];
    const float* b_hh0 = (const float*)d_in[4];
    const float* W_ih1 = (const float*)d_in[5];
    const float* W_hh1 = (const float*)d_in[6];
    const float* b_ih1 = (const float*)d_in[7];
    const float* b_hh1 = (const float*)d_in[8];
    const float* W_fc  = (const float*)d_in[9];
    const float* b_fc  = (const float*)d_in[10];
    float* out = (float*)d_out;

    dim3 grid(B_TOT / MT);      // 128 blocks, each owns 32 batch rows
    dim3 block(NTHREADS);       // 2 wave32s
    lstm2_wmma_kernel<<<grid, block, SMEM_BYTES, stream>>>(
        x, W_ih0, W_hh0, b_ih0, b_hh0, W_ih1, W_hh1, b_ih1, b_hh1, W_fc, b_fc, out);
}
